// guided_SpaSRModel_12283606466608
// MI455X (gfx1250) — matrix operand order repaired and checked by reading, not compiled, not run on hardware
//
#include <hip/hip_runtime.h>

// ---------------------------------------------------------------------------
// Types for CDNA5 WMMA (wave32): v_wmma_f32_16x16x32_bf16
// ---------------------------------------------------------------------------
typedef __attribute__((ext_vector_type(16))) __bf16 v16bf;
typedef __attribute__((ext_vector_type(8)))  __bf16 v8bf;
typedef __attribute__((ext_vector_type(8)))  float  v8f;
typedef __attribute__((ext_vector_type(4)))  float  v4f;

#define HWDIM 4096
#define CDIM  512
#define BDIM  4

__device__ inline v8f wmma_bf16(v16bf a, v16bf b, v8f c) {
  return __builtin_amdgcn_wmma_f32_16x16x32_bf16(false, a, false, b,
                                                 (short)0, c, false, false);
}

// A-frag (16x32, [m][k] row-major tile): lane L row m, elements are two
// contiguous 8-element runs at k = kofs + (L>>4)*8 and +16.  -> 2x ds_load_b128
__device__ inline v16bf frag_a_vec(const __bf16* base, int m, int stride,
                                   int lane, int kofs) {
  int kb = kofs + ((lane >> 4) << 3);
  union { v16bf v; v8bf h[2]; } u;
  u.h[0] = *(const v8bf*)(base + m * stride + kb);
  u.h[1] = *(const v8bf*)(base + m * stride + kb + 16);
  return u.v;
}
// B-frag (32x16) from a TRANSPOSED [n][k] tile: lane L row n, two contiguous
// 8-element runs at k = kofs + (L>>4)*8 and +16.  -> 2x ds_load_b128
__device__ inline v16bf frag_b_vec(const __bf16* baseT, int n, int stride,
                                   int lane, int kofs) {
  int kb = kofs + ((lane >> 4) << 3);
  union { v16bf v; v8bf h[2]; } u;
  u.h[0] = *(const v8bf*)(baseT + n * stride + kb);
  u.h[1] = *(const v8bf*)(baseT + n * stride + kb + 16);
  return u.v;
}

// ---------------------------------------------------------------------------
// K1: per-(b,c) mean / rstd for content & guidance (unbiased var, +1e-5)
// ---------------------------------------------------------------------------
__global__ __launch_bounds__(256) void inst_stats(
    const float* __restrict__ content, const float* __restrict__ guidance,
    float* __restrict__ meanC, float* __restrict__ rstdC,
    float* __restrict__ meanG, float* __restrict__ rstdG) {
  int id = blockIdx.x;
  int which = id >> 11;            // 0=content, 1=guidance  (B*C = 2048)
  int bc = id & 2047;
  const float* src = which ? guidance : content;
  long base = (long)bc * HWDIM;
  float s = 0.f, s2 = 0.f;
  for (int j = threadIdx.x; j < HWDIM; j += 256) {
    float v = src[base + j];
    s += v; s2 += v * v;
  }
  __shared__ float r1[256], r2[256];
  r1[threadIdx.x] = s; r2[threadIdx.x] = s2;
  __syncthreads();
  for (int st = 128; st > 0; st >>= 1) {
    if (threadIdx.x < st) {
      r1[threadIdx.x] += r1[threadIdx.x + st];
      r2[threadIdx.x] += r2[threadIdx.x + st];
    }
    __syncthreads();
  }
  if (threadIdx.x == 0) {
    float mu  = r1[0] * (1.f / HWDIM);
    float var = (r2[0] - (float)HWDIM * mu * mu) * (1.f / (HWDIM - 1)) + 1e-5f;
    float rs  = rsqrtf(var);
    if (which) { meanG[bc] = mu; rstdG[bc] = rs; }
    else       { meanC[bc] = mu; rstdC[bc] = rs; }
  }
}

// ---------------------------------------------------------------------------
// K2: normalize + bf16 + TRANSPOSE to [b][hw][c] via 64x64 LDS tile.
// Produces XcT = mvn(content)^T, XgT = mvn(guidance)^T, GdT = guidance^T.
// ---------------------------------------------------------------------------
__global__ __launch_bounds__(256) void pack_t(
    const float* __restrict__ content, const float* __restrict__ guidance,
    const float* __restrict__ meanC, const float* __restrict__ rstdC,
    const float* __restrict__ meanG, const float* __restrict__ rstdG,
    __bf16* __restrict__ XcT, __bf16* __restrict__ XgT, __bf16* __restrict__ GdT) {
  __shared__ __bf16 T[64 * 64];
  const int t = threadIdx.x;
  const int b = blockIdx.z, c0 = blockIdx.y * 64, h0 = blockIdx.x * 64;
  const int r = t >> 2, colb = (t & 3) * 16;       // read: c-row r, 16 hw cols
  const int orow = t >> 2, ocolb = (t & 3) * 16;   // write: hw-row, 16 c cols

  for (int tsel = 0; tsel < 3; ++tsel) {
    const float* src = (tsel == 0) ? content : guidance;
    float mu = 0.f, rs = 1.f;
    int cidx = b * CDIM + c0 + r;
    if (tsel == 0) { mu = meanC[cidx]; rs = rstdC[cidx]; }
    else if (tsel == 1) { mu = meanG[cidx]; rs = rstdG[cidx]; }
    long gbase = (long)cidx * HWDIM + h0 + colb;
#pragma unroll
    for (int u4 = 0; u4 < 4; ++u4) {
      v4f v = *(const v4f*)(src + gbase + u4 * 4);
#pragma unroll
      for (int u = 0; u < 4; ++u)
        T[(colb + u4 * 4 + u) * 64 + r] = (__bf16)((v[u] - mu) * rs);
    }
    __syncthreads();
    __bf16* dst = (tsel == 0) ? XcT : (tsel == 1) ? XgT : GdT;
    long obase = ((long)b * HWDIM + h0 + orow) * CDIM + c0 + ocolb;
    *(v8bf*)(dst + obase)     = *(const v8bf*)(T + orow * 64 + ocolb);
    *(v8bf*)(dst + obase + 8) = *(const v8bf*)(T + orow * 64 + ocolb + 8);
    __syncthreads();
  }
}

__global__ __launch_bounds__(256) void to_bf16(const float* __restrict__ src,
                                               __bf16* __restrict__ dst, int n) {
  int tid = (blockIdx.x * 256 + threadIdx.x) * 8;
  if (tid < n) {
    v8bf o;
#pragma unroll
    for (int u = 0; u < 8; ++u) o[u] = (__bf16)src[tid + u];
    *(v8bf*)(dst + tid) = o;
  }
}

// ---------------------------------------------------------------------------
// K4: bf16 GEMM: D[z][o][n] = sum_c W[o][c]*XT[z][n][c] + bias[o]
// WG tile 64(M) x 128(N); 8 waves of 32x32; K stepped 32 through LDS.
// As [m][k], Bs [n][k] (transposed) -> all fragment loads are ds_load_b128.
// ---------------------------------------------------------------------------
__global__ __launch_bounds__(256) void gemm_bf16(
    const __bf16* __restrict__ W, const __bf16* __restrict__ XT,
    const float* __restrict__ bias,
    float* __restrict__ outF, __bf16* __restrict__ outBT,
    __bf16* __restrict__ outBN, int M, int N, int K) {
  __shared__ __bf16 As[64 * 32];     // [m][k]
  __shared__ __bf16 Bs[128 * 32];    // [n][k]  (transposed)
  const int t = threadIdx.x, lane = t & 31, w = t >> 5;
  const int wm = w & 1, wn = w >> 1;
  const int m0 = blockIdx.y * 64, n0 = blockIdx.x * 128;
  const long z = blockIdx.z;

  v8f acc[2][2];
#pragma unroll
  for (int a0 = 0; a0 < 2; ++a0)
#pragma unroll
    for (int a1 = 0; a1 < 2; ++a1)
#pragma unroll
      for (int k = 0; k < 8; ++k) acc[a0][a1][k] = 0.f;

  for (int k0 = 0; k0 < K; k0 += 32) {
    {   // A tile 64x32: 8 contiguous bf16 per thread
      int r = t >> 2, c = (t & 3) * 8;
      *(v8bf*)(As + r * 32 + c) = *(const v8bf*)(W + (long)(m0 + r) * K + k0 + c);
    }
    {   // B tile 128(n)x32(k): 16 contiguous bf16 per thread
      int nl = t >> 1, kl = (t & 1) * 16;
      const __bf16* src = XT + (z * N + n0 + nl) * K + k0 + kl;
      *(v8bf*)(Bs + nl * 32 + kl)     = *(const v8bf*)(src);
      *(v8bf*)(Bs + nl * 32 + kl + 8) = *(const v8bf*)(src + 8);
    }
    if (k0 + 32 < K)                 // -> global_prefetch_b8
      __builtin_prefetch(XT + (z * N + n0 + (t >> 1)) * K + k0 + 32, 0, 1);
    __syncthreads();

    v16bf af[2], bfv[2];
#pragma unroll
    for (int mt = 0; mt < 2; ++mt)
      af[mt] = frag_a_vec(As, wm * 32 + mt * 16 + (lane & 15), 32, lane, 0);
#pragma unroll
    for (int nt = 0; nt < 2; ++nt)
      bfv[nt] = frag_b_vec(Bs, wn * 32 + nt * 16 + (lane & 15), 32, lane, 0);
#pragma unroll
    for (int mt = 0; mt < 2; ++mt)
#pragma unroll
      for (int nt = 0; nt < 2; ++nt)
        acc[mt][nt] = wmma_bf16(af[mt], bfv[nt], acc[mt][nt]);
    __syncthreads();
  }

  // epilogue: D layout lane L: n = L&15 ; m = vgpr + (L>>4)*8
#pragma unroll
  for (int mt = 0; mt < 2; ++mt)
#pragma unroll
    for (int nt = 0; nt < 2; ++nt)
#pragma unroll
      for (int v = 0; v < 8; ++v) {
        int o = m0 + wm * 32 + mt * 16 + v + ((lane >> 4) << 3);
        int n = n0 + wn * 32 + nt * 16 + (lane & 15);
        float val = acc[mt][nt][v] + bias[o];
        if (outF)  outF[(z * M + o) * (long)N + n] = val;
        if (outBN) outBN[(z * M + o) * (long)N + n] = (__bf16)val;
        if (outBT) outBT[(z * N + n) * (long)M + o] = (__bf16)val;
      }
}

// ---------------------------------------------------------------------------
// K5: single-pass flash attention.
//  Y[b,c,i] = (1/l_i) * sum_j exp(S[i,j]-m_i) * H[b,c,j],
//  S[i,j]  = sum_c F[b,c,i]*G1[b,c,j], online softmax over j.
// Inputs: FT/G1T transposed [b][hw][c] (Q rows / K rows), Hb normal [b][c][hw].
// ---------------------------------------------------------------------------
__global__ __launch_bounds__(256) void flash_attn(
    const __bf16* __restrict__ FT, const __bf16* __restrict__ KT,
    const __bf16* __restrict__ Hv, float* __restrict__ Y) {
  __shared__ __align__(16) unsigned char smem[16384 + 16384 + 8192 + 768];
  __bf16* stage = (__bf16*)smem;                 // Q:[i64][c128] K:[j64][c128] H:[c128][j64]
  float*  Sb    = (float*)(smem + 16384);        // [i64][j64]
  __bf16* P     = (__bf16*)(smem + 32768);       // [i64][j64] row-major
  float*  mrow  = (float*)(smem + 40960);
  float*  lrow  = mrow + 64;
  float*  srow  = lrow + 64;

  const int t = threadIdx.x, lane = t & 31, w = t >> 5;
  const int b = blockIdx.y;
  const int i0 = blockIdx.x * 64;
  const long baseB = (long)b * CDIM * HWDIM;     // for Hb / Y ([c][hw])
  const long baseT = (long)b * HWDIM * CDIM;     // for FT / KT ([hw][c])
  const int it_s = w & 3;        // score i-tile of this wave
  const int jhalf = w >> 2;      // score j-tile pair {2*jhalf, 2*jhalf+1}

  if (t < 64) { mrow[t] = -1e30f; lrow[t] = 0.f; }

  v8f acc[4][4];                 // [c-chunk][i-tile] : wave's 64c x 64i slice
#pragma unroll
  for (int a0 = 0; a0 < 4; ++a0)
#pragma unroll
    for (int a1 = 0; a1 < 4; ++a1)
#pragma unroll
      for (int k = 0; k < 8; ++k) acc[a0][a1][k] = 0.f;

  // ---- preload Q fragments (16 ksteps covering C=512) ----
  v16bf qf[16];
  for (int cc = 0; cc < 4; ++cc) {
#pragma unroll
    for (int e8 = 0; e8 < 4; ++e8) {             // Qs[i64][c128], 32 bf16/thread
      int lin = t * 32 + e8 * 8;
      int ii = lin >> 7, ci = lin & 127;
      *(v8bf*)(stage + lin) =
          *(const v8bf*)(FT + baseT + (long)(i0 + ii) * CDIM + cc * 128 + ci);
    }
    __syncthreads();
#pragma unroll
    for (int ks = 0; ks < 4; ++ks)
      qf[cc * 4 + ks] = frag_a_vec(stage, it_s * 16 + (lane & 15), 128, lane, ks * 32);
    __syncthreads();
  }

  // ---- stream over key/value j-tiles ----
  for (int j0 = 0; j0 < HWDIM; j0 += 64) {
    v8f sacc[2];
#pragma unroll
    for (int jt = 0; jt < 2; ++jt)
#pragma unroll
      for (int k = 0; k < 8; ++k) sacc[jt][k] = 0.f;

    for (int cc = 0; cc < 4; ++cc) {
#pragma unroll
      for (int e8 = 0; e8 < 4; ++e8) {           // Ks[j64][c128]
        int lin = t * 32 + e8 * 8;
        int jj = lin >> 7, ci = lin & 127;
        *(v8bf*)(stage + lin) =
            *(const v8bf*)(KT + baseT + (long)(j0 + jj) * CDIM + cc * 128 + ci);
      }
      __syncthreads();
#pragma unroll
      for (int ks = 0; ks < 4; ++ks)
#pragma unroll
        for (int jt = 0; jt < 2; ++jt) {
          v16bf bfv = frag_b_vec(stage, (jhalf * 2 + jt) * 16 + (lane & 15),
                                 128, lane, ks * 32);
          sacc[jt] = wmma_bf16(qf[cc * 4 + ks], bfv, sacc[jt]);
        }
      __syncthreads();
    }

    // write S tiles to LDS
#pragma unroll
    for (int jt = 0; jt < 2; ++jt)
#pragma unroll
      for (int v = 0; v < 8; ++v) {
        int r = it_s * 16 + v + ((lane >> 4) << 3);
        int cj = (jhalf * 2 + jt) * 16 + (lane & 15);
        Sb[r * 64 + cj] = sacc[jt][v];
      }
    __syncthreads();

    // online softmax per query row (threads 0..63 each own a row)
    if (t < 64) {
      float mo = mrow[t], mx = mo;
      for (int j = 0; j < 64; ++j) mx = fmaxf(mx, Sb[t * 64 + j]);
      float sc = __expf(mo - mx);
      float sum = 0.f;
#pragma unroll
      for (int j8 = 0; j8 < 64; j8 += 8) {
        v8bf pv;
#pragma unroll
        for (int u = 0; u < 8; ++u) {
          float p = __expf(Sb[t * 64 + j8 + u] - mx);
          sum += p;
          pv[u] = (__bf16)p;
        }
        *(v8bf*)(P + t * 64 + j8) = pv;          // P[i][j] row-major
      }
      mrow[t] = mx;
      lrow[t] = lrow[t] * sc + sum;
      srow[t] = sc;
    }
    __syncthreads();

    // rescale running accumulators (column i = itile*16 + lane&15)
#pragma unroll
    for (int itile = 0; itile < 4; ++itile) {
      float s = srow[itile * 16 + (lane & 15)];
#pragma unroll
      for (int cc = 0; cc < 4; ++cc)
#pragma unroll
        for (int v = 0; v < 8; ++v) acc[cc][itile][v] *= s;
    }

    // value stage: O[c,i] += H[c,j] * P[i,j], c streamed in 4 chunks of 128
    for (int cc = 0; cc < 4; ++cc) {
      __syncthreads();
#pragma unroll
      for (int e8 = 0; e8 < 4; ++e8) {           // Hs[c128][j64]
        int lin = t * 32 + e8 * 8;
        int ci = lin >> 6, jj = lin & 63;
        *(v8bf*)(stage + lin) =
            *(const v8bf*)(Hv + baseB + (long)(cc * 128 + ci) * HWDIM + j0 + jj);
      }
      __syncthreads();
#pragma unroll
      for (int ks = 0; ks < 2; ++ks) {
        v16bf af = frag_a_vec(stage, w * 16 + (lane & 15), 64, lane, ks * 32);
#pragma unroll
        for (int itile = 0; itile < 4; ++itile) {
          v16bf bfv = frag_b_vec(P, itile * 16 + (lane & 15), 64, lane, ks * 32);
          acc[cc][itile] = wmma_bf16(af, bfv, acc[cc][itile]);
        }
      }
    }
    __syncthreads();
  }

  // finalize: divide by row sums, write Y[b][c][i]
#pragma unroll
  for (int cc = 0; cc < 4; ++cc)
#pragma unroll
    for (int itile = 0; itile < 4; ++itile) {
      float inv = 1.f / lrow[itile * 16 + (lane & 15)];
#pragma unroll
      for (int v = 0; v < 8; ++v) {
        int c = cc * 128 + w * 16 + v + ((lane >> 4) << 3);
        int i = i0 + itile * 16 + (lane & 15);
        Y[baseB + (long)c * HWDIM + i] = acc[cc][itile][v] * inv;
      }
    }
}

// ---------------------------------------------------------------------------
// K6: in-place softmax over i (4096) per (b,c) row
// ---------------------------------------------------------------------------
__global__ __launch_bounds__(256) void softmax_rows(float* __restrict__ Y) {
  __shared__ float red[256];
  long base = (long)blockIdx.x * HWDIM;
  float mx = -1e30f;
  for (int j = threadIdx.x; j < HWDIM; j += 256) mx = fmaxf(mx, Y[base + j]);
  red[threadIdx.x] = mx; __syncthreads();
  for (int s = 128; s > 0; s >>= 1) {
    if (threadIdx.x < s) red[threadIdx.x] = fmaxf(red[threadIdx.x], red[threadIdx.x + s]);
    __syncthreads();
  }
  mx = red[0]; __syncthreads();
  float sum = 0.f;
  for (int j = threadIdx.x; j < HWDIM; j += 256) sum += __expf(Y[base + j] - mx);
  red[threadIdx.x] = sum; __syncthreads();
  for (int s = 128; s > 0; s >>= 1) {
    if (threadIdx.x < s) red[threadIdx.x] += red[threadIdx.x + s];
    __syncthreads();
  }
  float inv = 1.f / red[0];
  for (int j = threadIdx.x; j < HWDIM; j += 256)
    Y[base + j] = __expf(Y[base + j] - mx) * inv;
}

// ---------------------------------------------------------------------------
// K7: gamma/beta = weighted channel sums of G2
// ---------------------------------------------------------------------------
__global__ __launch_bounds__(256) void gamma_beta(
    const float* __restrict__ G2, const float* __restrict__ wg,
    const float* __restrict__ wb, float* __restrict__ gamma,
    float* __restrict__ beta) {
  int tid = blockIdx.x * 256 + threadIdx.x;     // B*HW threads
  int b = tid >> 12, i = tid & 4095;
  long base = (long)b * CDIM * HWDIM + i;
  float g = 0.f, be = 0.f;
  for (int c = 0; c < CDIM; ++c) {
    float v = G2[base + (long)c * HWDIM];
    g  += wg[c] * v;
    be += wb[c] * v;
  }
  gamma[tid] = g;
  beta[tid]  = be;
}

// ---------------------------------------------------------------------------
// K8: Out[b,c,h,w] = sum_k G1[b,c,h,k]*gamma[b,k,w] + beta[b,h,w]
// Thread order (b,hw,c) so the bf16 output is written TRANSPOSED ([b][hw][c],
// contiguous) ready to be the B-side of the final GEMM.
// ---------------------------------------------------------------------------
__global__ __launch_bounds__(256) void out_mat(
    const float* __restrict__ G1f, const float* __restrict__ gamma,
    const float* __restrict__ beta, __bf16* __restrict__ OutT) {
  long tid = (long)blockIdx.x * 256 + threadIdx.x;  // (b,hw,c) order
  int c  = (int)(tid & 511);
  int hw = (int)((tid >> 9) & 4095);
  int b  = (int)(tid >> 21);
  int h = hw >> 6, wc = hw & 63;
  float acc = beta[((long)b << 12) + hw];
  const float* g1row = G1f + ((long)(b * CDIM + c)) * HWDIM + (h << 6);
  const float* grow  = gamma + ((long)b << 12);
  for (int k = 0; k < 64; ++k)
    acc += g1row[k] * grow[(k << 6) + wc];
  OutT[tid] = (__bf16)acc;
}

// ---------------------------------------------------------------------------
// Orchestration
// ---------------------------------------------------------------------------
extern "C" void kernel_launch(void* const* d_in, const int* in_sizes, int n_in,
                              void* d_out, int out_size, void* d_ws, size_t ws_size,
                              hipStream_t stream) {
  const float* content  = (const float*)d_in[0];
  const float* guidance = (const float*)d_in[1];
  const float* Wg1  = (const float*)d_in[2];
  const float* bg1  = (const float*)d_in[3];
  const float* Wf   = (const float*)d_in[4];
  const float* bff  = (const float*)d_in[5];
  const float* Wh   = (const float*)d_in[6];
  const float* bh   = (const float*)d_in[7];
  const float* wgm  = (const float*)d_in[8];
  const float* wbt  = (const float*)d_in[9];
  const float* Wout = (const float*)d_in[10];
  const float* bout = (const float*)d_in[11];
  float* out = (float*)d_out;

  char* ws = (char*)d_ws;
  size_t off = 0;
  auto alloc = [&](size_t bytes) -> void* {
    void* p = (void*)(ws + off);
    off += (bytes + 255) & ~(size_t)255;
    return p;
  };

  const size_t NELEM = (size_t)BDIM * CDIM * HWDIM;     // 8388608
  float*  meanC = (float*)alloc(BDIM * CDIM * 4);
  float*  rstdC = (float*)alloc(BDIM * CDIM * 4);
  float*  meanG = (float*)alloc(BDIM * CDIM * 4);
  float*  rstdG = (float*)alloc(BDIM * CDIM * 4);
  __bf16* XcT   = (__bf16*)alloc(NELEM * 2);            // later reused as OutT
  __bf16* XgT   = (__bf16*)alloc(NELEM * 2);            // XgT+GdT later reused as Y
  __bf16* GdT   = (__bf16*)alloc(NELEM * 2);
  __bf16* Wg1b  = (__bf16*)alloc(CDIM * CDIM * 2);
  __bf16* Wfb   = (__bf16*)alloc(CDIM * CDIM * 2);
  __bf16* Whb   = (__bf16*)alloc(CDIM * CDIM * 2);
  __bf16* Woutb = (__bf16*)alloc(CDIM * CDIM * 2);
  float*  G1f   = (float*)alloc(NELEM * 4);
  __bf16* G1T   = (__bf16*)alloc(NELEM * 2);
  __bf16* FT    = (__bf16*)alloc(NELEM * 2);
  __bf16* Hb    = (__bf16*)alloc(NELEM * 2);
  float*  gam   = (float*)alloc(BDIM * HWDIM * 4);
  float*  bet   = (float*)alloc(BDIM * HWDIM * 4);
  (void)ws_size; (void)in_sizes; (void)n_in; (void)out_size;

  float*  Y    = (float*)XgT;   // 33.5MB overlay: XgT+GdT, both dead by then
  __bf16* OutT = XcT;           // XcT dead after G1 GEMM

  // 1) instance-norm statistics
  inst_stats<<<dim3(2 * BDIM * CDIM), dim3(256), 0, stream>>>(
      content, guidance, meanC, rstdC, meanG, rstdG);
  // 2) normalize + bf16 + transpose pack
  pack_t<<<dim3(HWDIM / 64, CDIM / 64, BDIM), dim3(256), 0, stream>>>(
      content, guidance, meanC, rstdC, meanG, rstdG, XcT, XgT, GdT);
  // 3) weights -> bf16
  to_bf16<<<dim3(CDIM * CDIM / 2048), dim3(256), 0, stream>>>(Wg1, Wg1b, CDIM * CDIM);
  to_bf16<<<dim3(CDIM * CDIM / 2048), dim3(256), 0, stream>>>(Wf,  Wfb,  CDIM * CDIM);
  to_bf16<<<dim3(CDIM * CDIM / 2048), dim3(256), 0, stream>>>(Wh,  Whb,  CDIM * CDIM);
  to_bf16<<<dim3(CDIM * CDIM / 2048), dim3(256), 0, stream>>>(Wout, Woutb, CDIM * CDIM);
  // 4) three 1x1 convs as bf16 WMMA GEMMs
  dim3 ggrid(HWDIM / 128, CDIM / 64, BDIM);
  gemm_bf16<<<ggrid, dim3(256), 0, stream>>>(Wg1b, XcT, bg1, G1f, G1T, nullptr,
                                             CDIM, HWDIM, CDIM);
  gemm_bf16<<<ggrid, dim3(256), 0, stream>>>(Wfb,  XgT, bff, nullptr, FT, nullptr,
                                             CDIM, HWDIM, CDIM);
  gemm_bf16<<<ggrid, dim3(256), 0, stream>>>(Whb,  GdT, bh, nullptr, nullptr, Hb,
                                             CDIM, HWDIM, CDIM);
  // 5) fused flash attention (row softmax over j + value accumulation)
  flash_attn<<<dim3(HWDIM / 64, BDIM), dim3(256), 0, stream>>>(FT, G1T, Hb, Y);
  // 6) softmax over i per (b,c)
  softmax_rows<<<dim3(BDIM * CDIM), dim3(256), 0, stream>>>(Y);
  // 7) gamma / beta channel sums
  gamma_beta<<<dim3(BDIM * HWDIM / 256), dim3(256), 0, stream>>>(Y, wgm, wbt, gam, bet);
  // 8) Out = G1 @ gamma + beta -> transposed bf16
  out_mat<<<dim3(NELEM / 256), dim3(256), 0, stream>>>(G1f, gam, bet, OutT);
  // 9) final 1x1 conv -> d_out (f32)
  gemm_bf16<<<ggrid, dim3(256), 0, stream>>>(Woutb, OutT, bout, out, nullptr, nullptr,
                                             CDIM, HWDIM, CDIM);
}
